// GraphAttentionNetwork_47132971106393
// MI455X (gfx1250) — matrix operand order repaired and checked by reading, not compiled, not run on hardware
//
#include <hip/hip_runtime.h>
#include <hip/hip_bf16.h>

typedef __attribute__((ext_vector_type(16))) _Float16 v16h;
typedef __attribute__((ext_vector_type(8)))  _Float16 v8h;
typedef __attribute__((ext_vector_type(4)))  _Float16 v4h;
typedef __attribute__((ext_vector_type(8)))  float    v8f;

#define NNODES 50000
#define NEDGES 800000
#define DIM    128
#define HEADS  8
#define UDIM   32
#define HU     256
#define OUTD   64
#define SKPAD  8      // LDS row pad (halves): 16B => conflict-free b128 reads

// ---------------------------------------------------------------------------
// WMMA GEMM: C[M,Ncol] = act(A[M,K] @ B + bias)
//   A:  f32 row-major [M,K];  Bt: f16 [Ncol][K] (pre-transposed in workspace)
// Grid: blockIdx.y = 16-row tile, (blockIdx.x * wavesPerBlock + wave) = col tile.
// Block cooperatively stages its A tile into LDS as f16 (one conversion pass),
// then each wave's hot loop is 2x ds_load_b128 + 2x global_load_b128 + 1 WMMA.
// No divergent exits: grid covers matrices exactly; EXEC all-ones at WMMA.
// Fragment layouts used here (contiguous K, not ISA table order; A: lane%16=row, K=16h+e;
// B: lane%16=col, K=16h+e; D: VGPR j -> row j / j+8).
// ---------------------------------------------------------------------------
__global__ void gemm_wmma_f16(const float* __restrict__ A,
                              const _Float16* __restrict__ Bt,
                              const float* __restrict__ bias,
                              float* __restrict__ C,
                              int K, int Ncol, int do_relu)
{
    __shared__ __align__(32) _Float16 As[16 * (256 + SKPAD)];
    const int SK   = K + SKPAD;
    const int lane = threadIdx.x & 31;
    const int wave = threadIdx.x >> 5;
    const int m0   = blockIdx.y << 4;
    const int n0   = (blockIdx.x * (blockDim.x >> 5) + wave) << 4;

    // ---- cooperative stage: 16 x K f32 tile -> f16 in LDS (padded rows) ----
    const int cpr = K >> 2;                       // float4 chunks per row
    for (int cidx = threadIdx.x; cidx < 16 * cpr; cidx += blockDim.x) {
        int r = cidx / cpr;
        int o = (cidx - r * cpr) << 2;
        const float4 g = *(const float4*)(A + (size_t)(m0 + r) * K + o);
        v4h h4;
        h4[0] = (_Float16)g.x; h4[1] = (_Float16)g.y;
        h4[2] = (_Float16)g.z; h4[3] = (_Float16)g.w;
        *(v4h*)(As + r * SK + o) = h4;
    }
    __syncthreads();

    const int half = lane >> 4;
    const int l15  = lane & 15;
    const _Float16* ar = As + l15 * SK + 16 * half;                // LDS row
    const _Float16* bp = Bt + (size_t)(n0 + l15) * K + 16 * half;  // global col

    v8f c = {};
#pragma unroll 2
    for (int k0 = 0; k0 < K; k0 += 32) {
        const v8h lo = *(const v8h*)(ar + k0);        // K = 16h + 0..7
        const v8h hi = *(const v8h*)(ar + k0 + 8);    // K = 16h + 8..15
        const v16h a = __builtin_shufflevector(lo, hi,
                          0,1,2,3,4,5,6,7,8,9,10,11,12,13,14,15);
        const v16h b = *(const v16h*)(bp + k0);       // K = 16h + 0..15
        c = __builtin_amdgcn_wmma_f32_16x16x32_f16(
                false, a, false, b, (short)0, c, false, false);
    }

    const float bv = bias ? bias[n0 + l15] : 0.0f;
#pragma unroll
    for (int j = 0; j < 8; ++j) {
        float v = c[j] + bv;
        if (do_relu) v = v > 0.0f ? v : 0.0f;
        C[(size_t)(m0 + j + 8 * half) * Ncol + (n0 + l15)] = v;
    }
}

// B[K,Ncol] f32 row-major  ->  Bt[Ncol,K] f16
__global__ void transpose_to_f16(const float* __restrict__ B,
                                 _Float16* __restrict__ Bt, int K, int Ncol)
{
    int t = blockIdx.x * blockDim.x + threadIdx.x;
    if (t >= K * Ncol) return;
    int cidx = t / K, k = t % K;
    Bt[t] = (_Float16)B[(size_t)k * Ncol + cidx];
}

// W_att[l] (H,HU,U) f32 -> Bt[HU,HU] f16 with col c = head*U+u
__global__ void repack_watt_f16(const float* __restrict__ W,
                                _Float16* __restrict__ Bt)
{
    int t = blockIdx.x * blockDim.x + threadIdx.x;
    if (t >= HU * HU) return;
    int cidx = t / HU, k = t % HU;
    int head = cidx >> 5, u = cidx & 31;
    Bt[t] = (_Float16)W[((size_t)head * HU + k) * UDIM + u];
}

// Per (node, head) attention scores: s_src = h . a[:U], s_dst = h . a[U:]
__global__ void head_scores(const float* __restrict__ h,
                            const float* __restrict__ a_l,
                            float* __restrict__ ssrc, float* __restrict__ sdst)
{
    int t = blockIdx.x * blockDim.x + threadIdx.x;
    if (t >= NNODES * HEADS) return;
    int n = t / HEADS, hd = t % HEADS;
    const float* hp = h + (size_t)n * HU + hd * UDIM;
    const float* as = a_l + hd * (2 * UDIM);
    float s0 = 0.f, s1 = 0.f;
#pragma unroll 8
    for (int u = 0; u < UDIM; ++u) {
        float hv = hp[u];
        s0 += hv * as[u];
        s1 += hv * as[UDIM + u];
    }
    ssrc[hd * NNODES + n] = s0;
    sdst[hd * NNODES + n] = s1;
}

__device__ __forceinline__ float edge_e_val(float s)
{
    float v = s > 0.f ? s : 0.2f * s;          // leaky_relu(0.2)
    v = fminf(2.f, fmaxf(-2.f, v));            // clip
    return __expf(v);
}

// Pass 1: softmax denominators, segment-summed over src.
__global__ void edge_den(const int* __restrict__ edges,
                         const float* __restrict__ ssrc,
                         const float* __restrict__ sdst,
                         float* __restrict__ den)
{
    long long t = (long long)blockIdx.x * blockDim.x + threadIdx.x;
    if (t >= (long long)NEDGES * HEADS) return;
    int e  = (int)(t / HEADS);
    int hd = (int)(t % HEADS);
    int s  = edges[2 * e];
    int d  = edges[2 * e + 1];
    float ev = edge_e_val(ssrc[hd * NNODES + s] + sdst[hd * NNODES + d]);
    atomicAdd(&den[hd * NNODES + s], ev);
}

// Pass 2: agg[src] += alpha * h[dst]; one 256-thread block per edge.
__global__ void edge_agg(const int* __restrict__ edges,
                         const float* __restrict__ ssrc,
                         const float* __restrict__ sdst,
                         const float* __restrict__ den,
                         const float* __restrict__ h,
                         float* __restrict__ agg)
{
    int e  = blockIdx.x;
    int cc = threadIdx.x;                       // 0..255
    int hd = cc >> 5;
    int s  = edges[2 * e];
    int d  = edges[2 * e + 1];
    float ev    = edge_e_val(ssrc[hd * NNODES + s] + sdst[hd * NNODES + d]);
    float alpha = ev / den[hd * NNODES + s];
    atomicAdd(&agg[(size_t)s * HU + cc], alpha * h[(size_t)d * HU + cc]);
}

// x = relu(agg) + x  (residual)
__global__ void residual_relu(const float* __restrict__ agg, float* __restrict__ x)
{
    int t = blockIdx.x * blockDim.x + threadIdx.x;
    if (t >= NNODES * HU) return;
    float a = agg[t];
    x[t] = (a > 0.f ? a : 0.f) + x[t];
}

extern "C" void kernel_launch(void* const* d_in, const int* in_sizes, int n_in,
                              void* d_out, int out_size, void* d_ws, size_t ws_size,
                              hipStream_t stream)
{
    const float* node_states = (const float*)d_in[0];
    const int*   edges       = (const int*)  d_in[1];
    const float* W_pre       = (const float*)d_in[2];
    const float* b_pre       = (const float*)d_in[3];
    const float* W_att       = (const float*)d_in[4];
    const float* a_att       = (const float*)d_in[5];
    const float* W_out       = (const float*)d_in[6];
    const float* b_out       = (const float*)d_in[7];
    float* out = (float*)d_out;

    // ---- workspace carve-out ------------------------------------------------
    float* x    = (float*)d_ws;                     // N*HU
    float* hfe  = x    + (size_t)NNODES * HU;       // N*HU
    float* agg  = hfe  + (size_t)NNODES * HU;       // N*HU
    float* ssrc = agg  + (size_t)NNODES * HU;       // H*N
    float* sdst = ssrc + (size_t)HEADS * NNODES;    // H*N
    float* den  = sdst + (size_t)HEADS * NNODES;    // H*N
    // f16 transposed weights (32B-aligned: float region is a 32B multiple)
    _Float16* BtPre = (_Float16*)(den + (size_t)HEADS * NNODES); // [HU][DIM]
    _Float16* BtAtt = BtPre + (size_t)HU * DIM;                  // [HU][HU]
    _Float16* BtOut = BtAtt + (size_t)HU * HU;                   // [OUTD][HU]

    // ---- weight repacks (f32 -> transposed f16) -----------------------------
    transpose_to_f16<<<(DIM * HU + 255) / 256, 256, 0, stream>>>(W_pre, BtPre, DIM, HU);
    transpose_to_f16<<<(HU * OUTD + 255) / 256, 256, 0, stream>>>(W_out, BtOut, HU, OUTD);

    const int tilesM = NNODES / 16;                 // 3125

    // x = relu(node_states @ W_pre + b_pre)   [Ncol=256 -> 16 col tiles, 8 waves]
    gemm_wmma_f16<<<dim3(2, tilesM), 256, 0, stream>>>(
        node_states, BtPre, b_pre, x, DIM, HU, 1);

    for (int l = 0; l < 2; ++l) {
        const float* Wl = W_att + (size_t)l * HEADS * HU * UDIM;
        const float* al = a_att + (size_t)l * HEADS * 2 * UDIM;

        repack_watt_f16<<<(HU * HU + 255) / 256, 256, 0, stream>>>(Wl, BtAtt);

        // h = x @ W_blk  (per-head projections concatenated)
        gemm_wmma_f16<<<dim3(2, tilesM), 256, 0, stream>>>(
            x, BtAtt, nullptr, hfe, HU, HU, 0);

        head_scores<<<(NNODES * HEADS + 255) / 256, 256, 0, stream>>>(
            hfe, al, ssrc, sdst);

        hipMemsetAsync(den, 0, sizeof(float) * (size_t)HEADS * NNODES, stream);
        hipMemsetAsync(agg, 0, sizeof(float) * (size_t)NNODES * HU, stream);

        long long tot = (long long)NEDGES * HEADS;
        edge_den<<<(int)((tot + 255) / 256), 256, 0, stream>>>(
            edges, ssrc, sdst, den);

        edge_agg<<<NEDGES, 256, 0, stream>>>(
            edges, ssrc, sdst, den, hfe, agg);

        residual_relu<<<(NNODES * HU + 255) / 256, 256, 0, stream>>>(agg, x);
    }

    // out = x @ W_out + b_out   [Ncol=64 -> 4 col tiles, 4 waves per block]
    gemm_wmma_f16<<<dim3(1, tilesM), 128, 0, stream>>>(
        x, BtOut, b_out, out, HU, OUTD, 0);
}
